// Interaction_Net_23175643529761
// MI455X (gfx1250) — matrix operand order repaired and checked by reading, not compiled, not run on hardware
//
#include <hip/hip_runtime.h>
#include <hip/hip_bf16.h>
#include <math.h>

// Problem constants (match reference)
#define B_   8
#define F_   16
#define N_   64
#define D_   64
#define HID  512
#define CELL 2056
#define BUF  1680

typedef __bf16 bvec16 __attribute__((ext_vector_type(16)));
typedef float  fvec8  __attribute__((ext_vector_type(8)));

union Frag { bvec16 v; uint4 u[2]; };

__device__ __forceinline__ unsigned short f2bf(float f) {
    unsigned int u = __float_as_uint(f);
    u += 0x7FFFu + ((u >> 16) & 1u);       // round-to-nearest-even
    return (unsigned short)(u >> 16);
}
__device__ __forceinline__ float gelu_exact(float x) {
    return 0.5f * x * (1.0f + erff(x * 0.70710678118654752f));
}
__device__ __forceinline__ float elu_f(float x) {
    return x > 0.f ? x : (expf(x) - 1.f);
}

// ---------------------------------------------------------------------------
// Kernel 1: object buffer  z(B,F,N,D) -> transpose(B,N,D,F) -> reshape(.,1024)
//           -> Dense(1024->BUF) + ELU   (f32, streaming, one-time ~1.8 GFLOP)
// ---------------------------------------------------------------------------
__global__ __launch_bounds__(128) void k_zb(const float* __restrict__ z,
                                            const float* __restrict__ bufW,
                                            const float* __restrict__ bufb,
                                            float* __restrict__ zb) {
    int c  = blockIdx.x * 128 + threadIdx.x;
    int bn = blockIdx.y;                       // 0..511  (b*64+n)
    if (c >= BUF) return;
    int b = bn >> 6, n = bn & 63;
    float acc = bufb[c];
    for (int d = 0; d < D_; ++d)
        for (int f = 0; f < F_; ++f)
            acc += z[(((b * F_ + f) * N_ + n) * D_ + d)] * bufW[(d * F_ + f) * BUF + c];
    zb[bn * BUF + c] = elu_f(acc);
}

// ---------------------------------------------------------------------------
// Kernel 2: factored layer-1 projections (the key FLOP reduction):
//   P = cells @ rhoW1[:CELL]      Q = cells @ rhoW1[CELL:]
//   R = cells @ lamW1[:CELL]      S = z_b   @ lamW1[CELL:]
// ---------------------------------------------------------------------------
__global__ __launch_bounds__(256) void k_proj(const float* __restrict__ cells,
                                              const float* __restrict__ rhoW1,
                                              const float* __restrict__ lamW1,
                                              const float* __restrict__ zb,
                                              float* __restrict__ P, float* __restrict__ Q,
                                              float* __restrict__ R, float* __restrict__ S) {
    int lin = blockIdx.x * 256 + threadIdx.x;     // < 512*512
    int sel = blockIdx.y;
    int r = lin >> 9, h = lin & 511;
    const float* x; const float* w; int K; float* dst;
    if (sel == 0)      { x = cells + r * CELL; w = rhoW1;              K = CELL; dst = P; }
    else if (sel == 1) { x = cells + r * CELL; w = rhoW1 + CELL * HID; K = CELL; dst = Q; }
    else if (sel == 2) { x = cells + r * CELL; w = lamW1;              K = CELL; dst = R; }
    else               { x = zb    + r * BUF;  w = lamW1 + CELL * HID; K = BUF;  dst = S; }
    float acc = 0.f;
    for (int k = 0; k < K; ++k) acc += x[k] * w[k * HID + h];
    dst[lin] = acc;
}

// ---------------------------------------------------------------------------
// Kernel 3: pack 512x512 f32 weights into bf16 WMMA B-fragments.
// B-fragment (32x16, bf16): lane l -> n = l&15, kbase = (l>>4)*16,
// element e (0..15) -> k = kt*32 + kbase + e.  Stored so the main loop reads
// each lane's 16 bf16 as one contiguous 32-byte chunk (2x b128, coalesced).
// ---------------------------------------------------------------------------
__global__ __launch_bounds__(256) void k_packw(const float* __restrict__ w0,
                                               const float* __restrict__ w1,
                                               const float* __restrict__ w2,
                                               const float* __restrict__ w3,
                                               unsigned short* __restrict__ dst) {
    int p = blockIdx.x * 256 + threadIdx.x;       // < 512*512
    int m = blockIdx.y;                           // matrix 0..3
    const float* W = (m == 0) ? w0 : (m == 1) ? w1 : (m == 2) ? w2 : w3;
    int nt   = p >> 13;           // / (16*512)
    int rem  = p & 8191;
    int kt   = rem >> 9;
    int q    = rem & 511;
    int lane = q >> 4;
    int e    = q & 15;
    int k = kt * 32 + ((lane >> 4) << 4) + e;
    int n = nt * 16 + (lane & 15);
    dst[m * HID * HID + p] = f2bf(W[k * HID + n]);
}

// ---------------------------------------------------------------------------
// WMMA GEMM layer: (32 x 512) @ (512 x 512) + bias + GELU.
// Each wave owns a fixed m-tile (wave&1) and 8 n-tiles (nt = (wave>>1)+4s),
// holding 8 INDEPENDENT f32 accumulators so back-to-back WMMAs have no
// D->A/B hazard, and loading each A-fragment from LDS only once per K-step
// (8x fewer ds_load_b128 than the naive tile loop).
// MODE 0: store bf16 to dstB   MODE 1: store f32 to dstC   MODE 2: dstC += val
// ---------------------------------------------------------------------------
template <int MODE>
__device__ __forceinline__ void gemm32x512(const unsigned short* sIn,
                                           unsigned short* dstB, float* dstC,
                                           const unsigned short* __restrict__ W,
                                           const float* __restrict__ bias,
                                           int lane, int wave) {
    const int mt   = wave & 1;
    const int nt0  = wave >> 1;                 // n-tiles: nt0 + 4*s, s = 0..7
    const int row  = mt * 16 + (lane & 15);
    const int ksel = (lane >> 4) * 16;

    fvec8 c[8];
#pragma unroll
    for (int s = 0; s < 8; ++s) c[s] = (fvec8){0.f, 0.f, 0.f, 0.f, 0.f, 0.f, 0.f, 0.f};

#pragma unroll 2
    for (int kt = 0; kt < 16; ++kt) {
        // A (16x32 bf16) from LDS: two 16B ds_load_b128, reused by 8 WMMAs
        Frag a;
        a.u[0] = *reinterpret_cast<const uint4*>(&sIn[row * HID + kt * 32 + ksel]);
        a.u[1] = *reinterpret_cast<const uint4*>(&sIn[row * HID + kt * 32 + ksel + 8]);
#pragma unroll
        for (int s = 0; s < 8; ++s) {
            const int nt = nt0 + 4 * s;
            const unsigned short* wp = W + ((nt * 16 + kt) * 512 + lane * 16);
            Frag bf;
            const uint4* pw = reinterpret_cast<const uint4*>(wp);
            bf.u[0] = pw[0];
            bf.u[1] = pw[1];
            if (kt + 1 < 16)                       // stream next K-panel early
                __builtin_prefetch(wp + 512, 0, 1);
            c[s] = __builtin_amdgcn_wmma_f32_16x16x32_bf16(false, a.v, false, bf.v,
                                                           (short)0, c[s], false, false);
        }
    }

    const float bv_lane = (float)(lane & 15);   // (unused dummy to keep layout clear)
    (void)bv_lane;
    const int rbase = mt * 16 + ((lane >> 4) << 3);
#pragma unroll
    for (int s = 0; s < 8; ++s) {
        const int nt = nt0 + 4 * s;
        const int bn = nt * 16 + (lane & 15);
        const float bv = bias[bn];
#pragma unroll
        for (int r = 0; r < 8; ++r) {
            float val = gelu_exact(c[s][r] + bv);
            int orow = rbase + r;
            if (MODE == 0)       dstB[orow * HID + bn] = f2bf(val);
            else if (MODE == 1)  dstC[orow * HID + bn] = val;
            else                 dstC[orow * HID + bn] += val;
        }
    }
}

// ---------------------------------------------------------------------------
// Kernel 4: fused pairwise MLP layers 2+3 (both paths) + sum/max reduction.
// One block per (b,i); j processed in chunks of 32.  8192 WMMA / block.
// LDS: 32KB (A,bf16) + 32KB (B,bf16) + 64KB (C,f32) = 128KB -> 2 WG / WGP.
// ---------------------------------------------------------------------------
__global__ __launch_bounds__(256) void k_main(
        const float* __restrict__ P, const float* __restrict__ Q,
        const float* __restrict__ R, const float* __restrict__ S,
        const float* __restrict__ rb1, const float* __restrict__ rb2,
        const float* __restrict__ rb3, const float* __restrict__ lb1,
        const float* __restrict__ lb2, const float* __restrict__ lb3,
        const unsigned short* __restrict__ wr2, const unsigned short* __restrict__ wr3,
        const unsigned short* __restrict__ wl2, const unsigned short* __restrict__ wl3,
        float* __restrict__ out) {
    __shared__ __align__(16) unsigned short sA[32 * HID];
    __shared__ __align__(16) unsigned short sB[32 * HID];
    __shared__ __align__(16) float          sC[32 * HID];

    const int tid  = threadIdx.x;
    const int lane = tid & 31;
    const int wave = tid >> 5;
    const int bi   = blockIdx.x;            // b*64 + i
    const int b    = bi >> 6;
    const int h0   = tid * 2, h1 = tid * 2 + 1;

    float accS0 = 0.f, accS1 = 0.f, accM0 = -INFINITY, accM1 = -INFINITY;

    for (int jc = 0; jc < 2; ++jc) {
        // ---- rho path: X1 = gelu(P_i + Q_j + b1) as bf16 A-operand in LDS
        for (int t = tid; t < 32 * HID; t += 256) {
            int jj = t >> 9, h = t & 511;
            int j  = jc * 32 + jj;
            float v = P[bi * HID + h] + Q[(b * N_ + j) * HID + h] + rb1[h];
            sA[t] = f2bf(gelu_exact(v));
        }
        __syncthreads();
        gemm32x512<0>(sA, sB, sC, wr2, rb2, lane, wave);   // X2 = gelu(X1@W2+b2)
        __syncthreads();
        gemm32x512<1>(sB, sA, sC, wr3, rb3, lane, wave);   // C  = gelu(X2@W3+b3)
        __syncthreads();

        // ---- lambda path: Y1 = gelu(R_i + S_j + b1)
        for (int t = tid; t < 32 * HID; t += 256) {
            int jj = t >> 9, h = t & 511;
            int j  = jc * 32 + jj;
            float v = R[bi * HID + h] + S[(b * N_ + j) * HID + h] + lb1[h];
            sA[t] = f2bf(gelu_exact(v));
        }
        __syncthreads();
        gemm32x512<0>(sA, sB, sC, wl2, lb2, lane, wave);   // Y2
        __syncthreads();
        gemm32x512<2>(sB, sA, sC, wl3, lb3, lane, wave);   // C += gelu(Y2@W3+b3)
        __syncthreads();

        // ---- per-(b,i) reduction over this j-chunk (each thread owns 2 cols)
        for (int jj = 0; jj < 32; ++jj) {
            float v0 = sC[jj * HID + h0], v1 = sC[jj * HID + h1];
            accS0 += v0; accS1 += v1;
            accM0 = fmaxf(accM0, v0); accM1 = fmaxf(accM1, v1);
        }
        __syncthreads();
    }
    out[bi * 1024 + h0]       = accS0;
    out[bi * 1024 + h1]       = accS1;
    out[bi * 1024 + 512 + h0] = accM0;
    out[bi * 1024 + 512 + h1] = accM1;
}

// ---------------------------------------------------------------------------
extern "C" void kernel_launch(void* const* d_in, const int* in_sizes, int n_in,
                              void* d_out, int out_size, void* d_ws, size_t ws_size,
                              hipStream_t stream) {
    const float* z     = (const float*)d_in[0];
    const float* cells = (const float*)d_in[1];
    const float* rW1 = (const float*)d_in[2];  const float* rb1 = (const float*)d_in[3];
    const float* rW2 = (const float*)d_in[4];  const float* rb2 = (const float*)d_in[5];
    const float* rW3 = (const float*)d_in[6];  const float* rb3 = (const float*)d_in[7];
    const float* lW1 = (const float*)d_in[8];  const float* lb1 = (const float*)d_in[9];
    const float* lW2 = (const float*)d_in[10]; const float* lb2 = (const float*)d_in[11];
    const float* lW3 = (const float*)d_in[12]; const float* lb3 = (const float*)d_in[13];
    const float* bufW = (const float*)d_in[14];
    const float* bufb = (const float*)d_in[15];
    float* out = (float*)d_out;

    // workspace layout (16B-aligned slices)
    char* ws = (char*)d_ws;
    const size_t ZB_BYTES = (size_t)(B_ * N_) * BUF * sizeof(float);      // 3,440,640
    const size_t PR_BYTES = (size_t)(B_ * N_) * HID * sizeof(float);      // 1,048,576
    float* zb = (float*)(ws);
    float* P  = (float*)(ws + ZB_BYTES);
    float* Q  = (float*)(ws + ZB_BYTES + 1 * PR_BYTES);
    float* R  = (float*)(ws + ZB_BYTES + 2 * PR_BYTES);
    float* S  = (float*)(ws + ZB_BYTES + 3 * PR_BYTES);
    unsigned short* wpk = (unsigned short*)(ws + ZB_BYTES + 4 * PR_BYTES);
    unsigned short* wr2 = wpk + 0 * HID * HID;
    unsigned short* wr3 = wpk + 1 * HID * HID;
    unsigned short* wl2 = wpk + 2 * HID * HID;
    unsigned short* wl3 = wpk + 3 * HID * HID;

    k_zb  <<<dim3((BUF + 127) / 128, B_ * N_), 128, 0, stream>>>(z, bufW, bufb, zb);
    k_proj<<<dim3((B_ * N_ * HID) / 256, 4), 256, 0, stream>>>(cells, rW1, lW1, zb, P, Q, R, S);
    k_packw<<<dim3((HID * HID) / 256, 4), 256, 0, stream>>>(rW2, rW3, lW2, lW3, wpk);
    k_main<<<B_ * N_, 256, 0, stream>>>(P, Q, R, S, rb1, rb2, rb3, lb1, lb2, lb3,
                                        wr2, wr3, wl2, wl3, out);
}
